// SparseGlobalMamba_56556129354515
// MI455X (gfx1250) — matrix operand order repaired and checked by reading, not compiled, not run on hardware
//
#include <hip/hip_runtime.h>

// ---------------- problem constants (from reference) ----------------
#define B_      16
#define T_      4096
#define K_      1024
#define D_IN    512
#define D_MODEL 256
#define D_OUT   256
#define NL      2
#define D_STATE 16
#define D_CONV  4
#define D_INNER 512
#define DT_RANK 16
#define NBK     (B_ * K_)     // 16384 rows
#define SKEW    8             // bf16 elements of LDS row padding (16 bytes = 4 dwords)

// ---------------- types ----------------
typedef __bf16 bf16_t;
typedef bf16_t bf16x16 __attribute__((ext_vector_type(16)));
typedef bf16_t bf16x8  __attribute__((ext_vector_type(8)));
typedef float  f32x8   __attribute__((ext_vector_type(8)));

// Force the TDM path in the device pass: if the builtin is absent / has a
// different arity we want the hard compile error (feedback), not a silent
// fallback. Host pass parses the cooperative-load fallback instead.
#if defined(__AMDGCN__)
#define HAVE_TDM 1
#else
#define HAVE_TDM 0
#endif

__device__ __forceinline__ bf16_t to_bf16(float f) {
    unsigned int u = __builtin_bit_cast(unsigned int, f);
    unsigned int r = (u + 0x7FFFu + ((u >> 16) & 1u)) >> 16;   // round-to-nearest-even
    return __builtin_bit_cast(bf16_t, (unsigned short)r);
}
__device__ __forceinline__ float siluf(float v) { return v / (1.f + __expf(-v)); }

#if HAVE_TDM
typedef unsigned int u32x4 __attribute__((ext_vector_type(4)));
typedef int          i32x4 __attribute__((ext_vector_type(4)));
typedef int          i32x8 __attribute__((ext_vector_type(8)));
// TDM: load a contiguous 64-row x (Ka/2)-dword weight panel into LDS with a
// 4-dword pad after every row (bank-conflict skew).
// D# bitfields per CDNA5 ISA (§8.3 group0, §8.4 group1).
__device__ __forceinline__ void tdm_load_panel(const bf16_t* Wp, int Ka, unsigned lds_off) {
    unsigned long long ga = (unsigned long long)(__UINTPTR_TYPE__)Wp;
    unsigned dw   = (unsigned)Ka >> 1;                               // dwords per row
    unsigned padi = 29u - (unsigned)__builtin_clz((unsigned)Ka);     // log2(Ka)-2: pad each row
    u32x4 g0;
    g0.x = 1u;                                        // count=1, user descriptor
    g0.y = lds_off;                                   // lds_addr (bytes)
    g0.z = (unsigned)ga;                              // global_addr[31:0]
    g0.w = (unsigned)((ga >> 32) & 0x01FFFFFFu) | (2u << 30); // global_addr[56:32] | type=2
    i32x8 g1;
    g1[0] = (int)((2u << 16) | (1u << 20) | (padi << 22) | (3u << 25));
             // wg_mask=0 | data_size=4B | pad_enable | pad_interval | pad_amount=3 (4 dwords)
    g1[1] = (int)((dw & 0xFFFFu) << 16);              // tensor_dim0[15:0]
    g1[2] = (int)(((dw >> 16) & 0xFFFFu) | (64u << 16)); // tensor_dim0[31:16] | tensor_dim1=64
    g1[3] = (int)((dw & 0xFFFFu) << 16);              // tensor_dim1 hi=0 | tile_dim0=dw
    g1[4] = 64;                                       // tile_dim1=64, tile_dim2=0
    g1[5] = (int)dw;                                  // tensor_dim0_stride[31:0]
    g1[6] = 0;                                        // stride0 hi | dim1_stride lo
    g1[7] = 0;                                        // dim1_stride hi
    i32x4 z4; z4[0] = 0; z4[1] = 0; z4[2] = 0; z4[3] = 0;
#if __clang_major__ >= 23
    i32x8 z8; z8[0]=0; z8[1]=0; z8[2]=0; z8[3]=0; z8[4]=0; z8[5]=0; z8[6]=0; z8[7]=0;
    __builtin_amdgcn_tensor_load_to_lds(g0, g1, z4, z4, z8, 0);
#else
    __builtin_amdgcn_tensor_load_to_lds(g0, g1, z4, z4, 0);
#endif
}
#endif

// ---------------- generic f32 -> bf16 cast ----------------
__global__ void k_cast_bf16(const float* __restrict__ in, bf16_t* __restrict__ out, size_t n) {
    size_t i = (size_t)blockIdx.x * blockDim.x + threadIdx.x;
    size_t stride = (size_t)gridDim.x * blockDim.x;
    for (; i < n; i += stride) out[i] = to_bf16(in[i]);
}

// ---------------- pad W_xp: [NL][48][512] -> [NL][64][512] bf16 (rows 48..63 = 0) ----------------
__global__ void k_pad_xp(const float* __restrict__ W_xp, bf16_t* __restrict__ out) {
    int gid = blockIdx.x * blockDim.x + threadIdx.x;          // NL*64*512
    if (gid >= NL * 64 * D_INNER) return;
    int col = gid & (D_INNER - 1);
    int row = (gid >> 9) & 63;
    int l   = gid >> 15;
    float v = (row < 48) ? W_xp[((size_t)l * 48 + row) * D_INNER + col] : 0.f;
    out[gid] = to_bf16(v);
}

// ---------------- pad W_dt: [NL][512][16] -> [NL][512][32] bf16 (cols 16..31 = 0) ----------------
__global__ void k_pad_dt(const float* __restrict__ W_dt, bf16_t* __restrict__ out) {
    int gid = blockIdx.x * blockDim.x + threadIdx.x;          // NL*512*32
    if (gid >= NL * D_INNER * 32) return;
    int col = gid & 31;
    int row = (gid >> 5) & (D_INNER - 1);
    int l   = gid >> 14;
    float v = (col < DT_RANK) ? W_dt[((size_t)l * D_INNER + row) * DT_RANK + col] : 0.f;
    out[gid] = to_bf16(v);
}

// ---------------- gather keyframes + cast ----------------
__global__ void k_gather_cast(const float* __restrict__ feats, const int* __restrict__ idx,
                              bf16_t* __restrict__ out) {
    int gid = blockIdx.x * blockDim.x + threadIdx.x;          // NBK * (D_IN/4)
    if (gid >= NBK * (D_IN / 4)) return;
    int c   = gid & (D_IN / 4 - 1);
    int row = gid >> 7;
    int b   = row >> 10;
    int t   = idx[row];
    const float4 f = *(const float4*)(feats + ((size_t)b * T_ + t) * D_IN + c * 4);
    bf16_t* o = out + (size_t)row * D_IN + c * 4;
    o[0] = to_bf16(f.x); o[1] = to_bf16(f.y); o[2] = to_bf16(f.z); o[3] = to_bf16(f.w);
}

// ---------------- WMMA GEMM (all dims aligned: M%64==0, N%64==0, Ka%32==0, ldw==Ka) -------------
// C[MxN](f32) = act(A[MxKa](bf16) * W[NxKa]^T + bias)   [ACCUM: C += ...]
// Block = 128 threads = 4 waves; each wave owns a 16x64 tile (4x v_wmma_f32_16x16x32_bf16).
// The 64xKa weight panel is staged into LDS once per block by the Tensor Data Mover,
// with a 16B/row skew so the 16 lanes of each half span all 64 LDS banks.
template<int ACT, int ACCUM>
__global__ __launch_bounds__(128)
void k_gemm_bf16(const bf16_t* __restrict__ A, int lda,
                 const bf16_t* __restrict__ W,
                 float* __restrict__ C, int ldc, int Ka,
                 const float* __restrict__ bias) {
    extern __shared__ bf16_t smem[];                  // 64 * (Ka + SKEW) bf16
    const int lane = threadIdx.x & 31;
    const int wave = threadIdx.x >> 5;
    const int rowBase = (blockIdx.y * 4 + wave) * 16;
    const int colBase = blockIdx.x * 64;
    const int half = lane >> 4;     // 0 or 1
    const int l16  = lane & 15;
    const int ldsld = Ka + SKEW;                      // skewed LDS row stride (elements)

    const bf16_t* Wp = W + (size_t)colBase * Ka;      // contiguous 64-row panel
#if HAVE_TDM
    if (wave == 0) {
        tdm_load_panel(Wp, Ka, __builtin_amdgcn_groupstaticsize());
        __builtin_amdgcn_s_wait_tensorcnt(0);
    }
#else
    {
        const int rowvec = Ka / 8;                    // bf16x8 chunks per row
        for (int i = threadIdx.x; i < 64 * rowvec; i += 128) {
            int r = i / rowvec, c = i - r * rowvec;
            *(bf16x8*)(smem + (size_t)r * ldsld + c * 8) =
                ((const bf16x8*)(Wp + (size_t)r * Ka))[c];
        }
    }
#endif
    __syncthreads();

    f32x8 acc0 = {0,0,0,0,0,0,0,0}, acc1 = {0,0,0,0,0,0,0,0};
    f32x8 acc2 = {0,0,0,0,0,0,0,0}, acc3 = {0,0,0,0,0,0,0,0};

    const bf16_t* arow = A + (size_t)(rowBase + l16) * lda;
    for (int kk = 0; kk < Ka; kk += 32) {
        __builtin_prefetch(arow + kk + 256, 0, 3);    // global_prefetch_b8
        // A fragment: 16-bit 16x32 layout; lane half h holds K = h*8+{0..7} and 16+h*8+{0..7}
        bf16x8 lo = *(const bf16x8*)(arow + kk + half * 8);
        bf16x8 hi = *(const bf16x8*)(arow + kk + 16 + half * 8);
        bf16x16 a;
#pragma unroll
        for (int j = 0; j < 8; ++j) { a[j] = lo[j]; a[8 + j] = hi[j]; }
        // B fragments from LDS: lane = column, lane half h holds K = h*8+{0..7} and 16+h*8+{0..7}
        const bf16_t* brow = smem + (size_t)l16 * ldsld + kk + half * 8;
#define LOAD_B(BV, T)                                                           \
        bf16x16 BV;                                                            \
        {                                                                      \
            const bf16_t* bp = brow + (size_t)(T) * 16 * ldsld;                \
            bf16x8 blo = *(const bf16x8*)(bp);                                 \
            bf16x8 bhi = *(const bf16x8*)(bp + 16);                            \
            _Pragma("unroll")                                                  \
            for (int j = 0; j < 8; ++j) { BV[j] = blo[j]; BV[8 + j] = bhi[j]; }\
        }
        LOAD_B(b0, 0) LOAD_B(b1, 1) LOAD_B(b2, 2) LOAD_B(b3, 3)
#undef LOAD_B
        acc0 = __builtin_amdgcn_wmma_f32_16x16x32_bf16(false, a, false, b0, (short)0, acc0, false, false);
        acc1 = __builtin_amdgcn_wmma_f32_16x16x32_bf16(false, a, false, b1, (short)0, acc1, false, false);
        acc2 = __builtin_amdgcn_wmma_f32_16x16x32_bf16(false, a, false, b2, (short)0, acc2, false, false);
        acc3 = __builtin_amdgcn_wmma_f32_16x16x32_bf16(false, a, false, b3, (short)0, acc3, false, false);
    }
    // Epilogue: C/D layout — VGPR v: lanes 0-15 -> M=v, lanes 16-31 -> M=8+v; N = lane&15
#define EPILOGUE(ACCV, T)                                                        \
    {                                                                            \
        int n = colBase + (T) * 16 + l16;                                        \
        float bv = bias ? bias[n] : 0.f;                                         \
        _Pragma("unroll")                                                        \
        for (int v = 0; v < 8; ++v) {                                            \
            int m = rowBase + half * 8 + v;                                      \
            float val = ACCV[v] + bv;                                            \
            if (ACT == 1) val = (val > 20.f) ? val : log1pf(__expf(val));        \
            float* cp = C + (size_t)m * ldc + n;                                 \
            if (ACCUM) *cp += val; else *cp = val;                               \
        }                                                                        \
    }
    EPILOGUE(acc0, 0) EPILOGUE(acc1, 1) EPILOGUE(acc2, 2) EPILOGUE(acc3, 3)
#undef EPILOGUE
}

// ---------------- LayerNorm (D=256) + cast to bf16 ----------------
__global__ __launch_bounds__(256)
void k_layernorm_cast(const float* __restrict__ x, const float* __restrict__ g,
                      const float* __restrict__ b, bf16_t* __restrict__ out) {
    __shared__ float red[256];
    const int row = blockIdx.x, tid = threadIdx.x;
    float v = x[(size_t)row * D_MODEL + tid];
    red[tid] = v; __syncthreads();
    for (int s = 128; s > 0; s >>= 1) { if (tid < s) red[tid] += red[tid + s]; __syncthreads(); }
    float mu = red[0] / D_MODEL; __syncthreads();
    float d = v - mu;
    red[tid] = d * d; __syncthreads();
    for (int s = 128; s > 0; s >>= 1) { if (tid < s) red[tid] += red[tid + s]; __syncthreads(); }
    float r = rsqrtf(red[0] / D_MODEL + 1e-5f);
    out[(size_t)row * D_MODEL + tid] = to_bf16(d * r * g[tid] + b[tid]);
}

// ---------------- causal depthwise conv (k=4) + bias + SiLU ----------------
__global__ void k_conv_silu(const float* __restrict__ xp, const float* __restrict__ cw,
                            const float* __restrict__ cb, float* __restrict__ xc,
                            bf16_t* __restrict__ xcb) {
    int gid = blockIdx.x * blockDim.x + threadIdx.x;          // NBK * D_INNER
    if (gid >= NBK * D_INNER) return;
    int d   = gid & (D_INNER - 1);
    int row = gid >> 9;
    int k   = row & (K_ - 1);
    float acc = cb[d];
#pragma unroll
    for (int j = 0; j < D_CONV; ++j) {
        int kk = k - (D_CONV - 1) + j;
        if (kk >= 0) acc += cw[d * D_CONV + j] * xp[(size_t)(row - (D_CONV - 1) + j) * D_INNER + d];
    }
    float s = siluf(acc);
    xc[gid]  = s;
    xcb[gid] = to_bf16(s);
}

// ---------------- selective scan: thread per (b, d); x_dbl row stride = 64 ----------------
__global__ __launch_bounds__(256)
void k_scan(const float* __restrict__ dt, const float* __restrict__ xc,
            const float* __restrict__ xdbl, const float* __restrict__ A_log,
            const float* __restrict__ Dp, float* __restrict__ y) {
    const int d = blockIdx.x * blockDim.x + threadIdx.x;      // 0..511
    const int b = blockIdx.y;
    float Av[D_STATE];
#pragma unroll
    for (int s = 0; s < D_STATE; ++s) Av[s] = -__expf(A_log[d * D_STATE + s]);
    float h[D_STATE];
#pragma unroll
    for (int s = 0; s < D_STATE; ++s) h[s] = 0.f;
    const float Dpd = Dp[d];
    for (int k = 0; k < K_; ++k) {
        const size_t row = (size_t)b * K_ + k;
        const float dtv = dt[row * D_INNER + d];
        const float xcv = xc[row * D_INNER + d];
        const float* bd = xdbl + row * 64;                    // [0:16)=dt_low [16:32)=B [32:48)=C
        float yv = 0.f;
#pragma unroll
        for (int s = 0; s < D_STATE; ++s) {
            float dA = __expf(dtv * Av[s]);
            h[s] = dA * h[s] + dtv * bd[16 + s] * xcv;
            yv += h[s] * bd[32 + s];
        }
        y[row * D_INNER + d] = yv + Dpd * xcv;
    }
}

// ---------------- y * silu(z) -> bf16 ----------------
__global__ void k_gate_cast(const float* __restrict__ y, const float* __restrict__ z,
                            bf16_t* __restrict__ out, size_t n) {
    size_t i = (size_t)blockIdx.x * blockDim.x + threadIdx.x;
    size_t stride = (size_t)gridDim.x * blockDim.x;
    for (; i < n; i += stride) out[i] = to_bf16(y[i] * siluf(z[i]));
}

// ---------------- masked mean over K ----------------
__global__ __launch_bounds__(256)
void k_mean(const float* __restrict__ xo, const unsigned char* __restrict__ mask,
            float* __restrict__ out) {
    const int b = blockIdx.x, o = threadIdx.x;
    float sum = 0.f, cnt = 0.f;
    for (int k = 0; k < K_; ++k) {
        float m = mask[b * K_ + k] ? 0.f : 1.f;
        cnt += m;
        sum += m * xo[((size_t)b * K_ + k) * D_OUT + o];
    }
    if (cnt < 1.f) cnt = 1.f;
    out[b * D_OUT + o] = sum / cnt;
}

// ---------------- host side ----------------
static inline void launch_gemm(const bf16_t* A, int lda, const bf16_t* W,
                               float* C, int ldc, int M, int N, int Ka,
                               const float* bias, int act, int accum, hipStream_t s) {
    dim3 g(N / 64, M / 64);
    size_t shmem = (size_t)64 * (Ka + SKEW) * sizeof(bf16_t);
    if (act == 1)
        k_gemm_bf16<1, 0><<<g, 128, shmem, s>>>(A, lda, W, C, ldc, Ka, bias);
    else if (accum)
        k_gemm_bf16<0, 1><<<g, 128, shmem, s>>>(A, lda, W, C, ldc, Ka, bias);
    else
        k_gemm_bf16<0, 0><<<g, 128, shmem, s>>>(A, lda, W, C, ldc, Ka, bias);
}
static inline void launch_cast(const float* in, bf16_t* out, size_t n, hipStream_t s) {
    int blocks = (int)((n + 255) / 256); if (blocks > 65535) blocks = 65535;
    k_cast_bf16<<<blocks, 256, 0, s>>>(in, out, n);
}

extern "C" void kernel_launch(void* const* d_in, const int* in_sizes, int n_in,
                              void* d_out, int out_size, void* d_ws, size_t ws_size,
                              hipStream_t stream) {
    (void)in_sizes; (void)n_in; (void)out_size; (void)ws_size;
    const float* feats  = (const float*)d_in[0];
    const int*   idx    = (const int*)d_in[1];
    const unsigned char* mask = (const unsigned char*)d_in[2];
    const float* w_ip   = (const float*)d_in[3];
    const float* b_ip   = (const float*)d_in[4];
    const float* ln_g   = (const float*)d_in[5];
    const float* ln_b   = (const float*)d_in[6];
    const float* W_in   = (const float*)d_in[7];
    const float* conv_w = (const float*)d_in[8];
    const float* conv_b = (const float*)d_in[9];
    const float* W_xp   = (const float*)d_in[10];
    const float* W_dt   = (const float*)d_in[11];
    const float* b_dt   = (const float*)d_in[12];
    const float* A_log  = (const float*)d_in[13];
    const float* Dp     = (const float*)d_in[14];
    const float* W_out  = (const float*)d_in[15];
    const float* w_op   = (const float*)d_in[16];
    const float* b_op   = (const float*)d_in[17];
    float* out = (float*)d_out;

    // ---- workspace carve-out ----
    char* p = (char*)d_ws;
    auto alloc = [&](size_t bytes) { char* r = p; p += (bytes + 255) & ~(size_t)255; return r; };
    bf16_t* wb_ip  = (bf16_t*)alloc((size_t)D_MODEL * D_IN * 2);
    bf16_t* wb_in  = (bf16_t*)alloc((size_t)NL * 2 * D_INNER * D_MODEL * 2);
    bf16_t* wb_xpp = (bf16_t*)alloc((size_t)NL * 64 * D_INNER * 2);        // padded 48->64 rows
    bf16_t* wb_dtp = (bf16_t*)alloc((size_t)NL * D_INNER * 32 * 2);        // padded 16->32 cols
    bf16_t* wb_out = (bf16_t*)alloc((size_t)NL * D_MODEL * D_INNER * 2);
    bf16_t* wb_op  = (bf16_t*)alloc((size_t)D_OUT * D_MODEL * 2);
    float* x_f    = (float*)alloc((size_t)NBK * D_MODEL * 4);
    float* xp_f   = (float*)alloc((size_t)NBK * D_INNER * 4);   // reused as y after conv
    float* z_f    = (float*)alloc((size_t)NBK * D_INNER * 4);
    float* xc_f   = (float*)alloc((size_t)NBK * D_INNER * 4);
    float* xdbl_f = (float*)alloc((size_t)NBK * 64 * 4);        // ldc = 64 (cols 48..63 = 0)
    float* dt_f   = (float*)alloc((size_t)NBK * D_INNER * 4);
    float* xo_f   = (float*)alloc((size_t)NBK * D_OUT * 4);
    bf16_t* ab512 = (bf16_t*)alloc((size_t)NBK * D_INNER * 2);  // kf_bf / xc_bf / y2_bf
    bf16_t* ab256 = (bf16_t*)alloc((size_t)NBK * D_MODEL * 2);  // xn_bf / x_bf
    bf16_t* ab64  = (bf16_t*)alloc((size_t)NBK * 64 * 2);       // xdbl_bf (stride 64)

    // ---- 1) weights -> bf16 (row-major matches WMMA B-operand lane layout) ----
    launch_cast(w_ip,  wb_ip,  (size_t)D_MODEL * D_IN, stream);
    launch_cast(W_in,  wb_in,  (size_t)NL * 2 * D_INNER * D_MODEL, stream);
    k_pad_xp<<<NL * 64 * D_INNER / 256, 256, 0, stream>>>(W_xp, wb_xpp);
    k_pad_dt<<<NL * D_INNER * 32 / 256, 256, 0, stream>>>(W_dt, wb_dtp);
    launch_cast(W_out, wb_out, (size_t)NL * D_MODEL * D_INNER, stream);
    launch_cast(w_op,  wb_op,  (size_t)D_OUT * D_MODEL, stream);

    // ---- 2) gather keyframes + input projection ----
    k_gather_cast<<<NBK * (D_IN / 4) / 256, 256, 0, stream>>>(feats, idx, ab512);
    launch_gemm(ab512, D_IN, wb_ip, x_f, D_MODEL, NBK, D_MODEL, D_IN, b_ip, 0, 0, stream);

    // ---- 3) mamba layers ----
    for (int l = 0; l < NL; ++l) {
        k_layernorm_cast<<<NBK, 256, 0, stream>>>(x_f, ln_g + l * D_MODEL, ln_b + l * D_MODEL, ab256);
        const bf16_t* Wl = wb_in + (size_t)l * 2 * D_INNER * D_MODEL;
        launch_gemm(ab256, D_MODEL, Wl, xp_f, D_INNER, NBK, D_INNER, D_MODEL, nullptr, 0, 0, stream);
        launch_gemm(ab256, D_MODEL, Wl + (size_t)D_INNER * D_MODEL,
                    z_f, D_INNER, NBK, D_INNER, D_MODEL, nullptr, 0, 0, stream);
        k_conv_silu<<<NBK * D_INNER / 256, 256, 0, stream>>>(
            xp_f, conv_w + l * D_INNER * D_CONV, conv_b + l * D_INNER, xc_f, ab512);
        launch_gemm(ab512, D_INNER, wb_xpp + (size_t)l * 64 * D_INNER,
                    xdbl_f, 64, NBK, 64, D_INNER, nullptr, 0, 0, stream);
        launch_cast(xdbl_f, ab64, (size_t)NBK * 64, stream);
        // dt: Ka padded to 32 (cols 16..31 of A hit zero weight cols) -> aligned fast path
        launch_gemm(ab64, 64, wb_dtp + (size_t)l * D_INNER * 32,
                    dt_f, D_INNER, NBK, D_INNER, 32, b_dt + l * D_INNER, /*softplus*/1, 0, stream);
        k_scan<<<dim3(D_INNER / 256, B_), 256, 0, stream>>>(
            dt_f, xc_f, xdbl_f, A_log + (size_t)l * D_INNER * D_STATE, Dp + l * D_INNER, xp_f);
        {
            size_t n = (size_t)NBK * D_INNER;
            int blocks = (int)((n + 255) / 256); if (blocks > 65535) blocks = 65535;
            k_gate_cast<<<blocks, 256, 0, stream>>>(xp_f, z_f, ab512, n);
        }
        // residual folded in via accumulate epilogue: x += y2 @ W_out^T
        launch_gemm(ab512, D_INNER, wb_out + (size_t)l * D_MODEL * D_INNER,
                    x_f, D_MODEL, NBK, D_MODEL, D_INNER, nullptr, 0, /*accum*/1, stream);
    }

    // ---- 4) output projection + masked mean ----
    launch_cast(x_f, ab256, (size_t)NBK * D_MODEL, stream);
    launch_gemm(ab256, D_MODEL, wb_op, xo_f, D_OUT, NBK, D_OUT, D_MODEL, b_op, 0, 0, stream);
    k_mean<<<B_, D_OUT, 0, stream>>>(xo_f, mask, out);
}
